// GAT_75110388073023
// MI455X (gfx1250) — compile-verified, matrix operand order checked by reading, not run on hardware
//
#include <hip/hip_runtime.h>
#include <hip/hip_bf16.h>

#define NN    100000
#define F_INN 128
#define HID   64
#define NE    1600000
#define EPSA  1e-16f
#define SLOPE 0.2f

typedef __bf16 bf16_t;
typedef bf16_t v16bf __attribute__((ext_vector_type(16)));
typedef float  v8f   __attribute__((ext_vector_type(8)));

union FragU { v16bf v; uint4 q[2]; };

__device__ __forceinline__ unsigned short f2bf(float f) {
    unsigned u = __float_as_uint(f);
    unsigned r = u + 0x7FFFu + ((u >> 16) & 1u);   // RNE
    return (unsigned short)(r >> 16);
}

__device__ __forceinline__ void atomicMaxF(float* addr, float val) {
    if (val >= 0.0f) atomicMax((int*)addr, __float_as_int(val));
    else             atomicMin((unsigned int*)addr, __float_as_uint(val));
}

// ---------------- conversion / init kernels ----------------

__global__ void cvt_bf16_k(const float* __restrict__ in, unsigned short* __restrict__ out, int n) {
    int t = blockIdx.x * blockDim.x + threadIdx.x;
    if (t < n) out[t] = f2bf(in[t]);
}

// W[K,Ncol] (row-major) -> Wt[Ncol,K] bf16 (column of W contiguous)
__global__ void cvt_w_t_k(const float* __restrict__ W, unsigned short* __restrict__ Wt, int K, int Ncol) {
    int t = blockIdx.x * blockDim.x + threadIdx.x;
    if (t >= K * Ncol) return;
    int k = t / Ncol, c = t % Ncol;
    Wt[c * K + k] = f2bf(W[t]);
}

__global__ void bias_relu_bf16_k(const float* __restrict__ in, const float* __restrict__ b,
                                 unsigned short* __restrict__ out, int n) {
    int t = blockIdx.x * blockDim.x + threadIdx.x;
    if (t >= n) return;
    float v = in[t] + b[t & 63];
    out[t] = f2bf(v > 0.f ? v : 0.f);
}

__global__ void zero_f32_k(float* __restrict__ p, int n) {
    int t = blockIdx.x * blockDim.x + threadIdx.x;
    if (t < n) p[t] = 0.f;
}

// ---------------- WMMA GEMM: C[M,64] = A_bf16[M,K] @ Bt_bf16[64,K]^T ----------------

template <int K>
__global__ void gemm_bf16_wmma(const unsigned short* __restrict__ A,
                               const unsigned short* __restrict__ Bt,
                               float* __restrict__ C, int M) {
    const int wave = threadIdx.x >> 5;     // 8 waves / block (wave32)
    const int lane = threadIdx.x & 31;
    const int l16  = lane & 15;
    const int kh   = lane >> 4;            // 0: K{0..7,16..23}, 1: K{8..15,24..31}
    const int rowBase = blockIdx.x * 128 + wave * 16;

    int arow = rowBase + l16;
    if (arow > M - 1) arow = M - 1;        // clamp; stores are guarded
    const unsigned short* ap = A + (size_t)arow * K;

    v8f acc[4];
#pragma unroll
    for (int j = 0; j < 4; ++j) acc[j] = (v8f)0.0f;

#pragma unroll
    for (int kk = 0; kk < K; kk += 32) {
        const int kb = kk + kh * 8;
        FragU a;
        a.q[0] = *(const uint4*)(ap + kb);
        a.q[1] = *(const uint4*)(ap + kb + 16);
#pragma unroll
        for (int j = 0; j < 4; ++j) {
            const unsigned short* bp = Bt + (size_t)(j * 16 + l16) * K + kb;
            FragU b;
            b.q[0] = *(const uint4*)(bp);
            b.q[1] = *(const uint4*)(bp + 16);
            acc[j] = __builtin_amdgcn_wmma_f32_16x16x32_bf16(
                false, a.v, false, b.v, (short)0, acc[j], false, false);
        }
    }

#pragma unroll
    for (int j = 0; j < 4; ++j) {
        const int col = j * 16 + l16;
#pragma unroll
        for (int r = 0; r < 8; ++r) {
            const int row = rowBase + r + (kh ? 8 : 0);
            if (row < M) C[(size_t)row * 64 + col] = acc[j][r];
        }
    }
}

// ---------------- per-node alpha + softmax-state init ----------------

template <int H, int C>
__global__ void node_alpha_k(const float* __restrict__ h, const float* __restrict__ a_s,
                             const float* __restrict__ a_d, float* __restrict__ as,
                             float* __restrict__ ad, float* __restrict__ m,
                             float* __restrict__ den, int n_nodes) {
    int t = blockIdx.x * blockDim.x + threadIdx.x;
    if (t >= n_nodes * H) return;
    int n = t / H, hh = t % H;
    const float* row = h + (size_t)n * (H * C) + hh * C;
    float ss = 0.f, dd = 0.f;
#pragma unroll
    for (int c = 0; c < C; ++c) {
        float v = row[c];
        ss += v * a_s[hh * C + c];
        dd += v * a_d[hh * C + c];
    }
    as[t] = ss; ad[t] = dd;
    m[t]  = -__builtin_inff();
    den[t] = 0.f;
}

// ---------------- edge passes ----------------

template <int H>
__global__ void edge_max_k(const long long* __restrict__ ei, const float* __restrict__ as,
                           const float* __restrict__ ad, float* __restrict__ ebuf,
                           float* __restrict__ m, int E) {
    int e = blockIdx.x * blockDim.x + threadIdx.x;
    if (e >= E) return;
    __builtin_prefetch(ei + e + 16384, 0, 0);          // edge-index stream -> global_prefetch_b8
    int s = (int)ei[e];
    int d = (int)ei[E + e];
#pragma unroll
    for (int h = 0; h < H; ++h) {
        float v = as[s * H + h] + ad[d * H + h];
        v = v > 0.f ? v : SLOPE * v;                   // leaky relu
        ebuf[e * H + h] = v;
        atomicMaxF(&m[d * H + h], v);
    }
}

template <int H>
__global__ void edge_exp_k(const long long* __restrict__ ei, float* __restrict__ ebuf,
                           const float* __restrict__ m, float* __restrict__ den, int E) {
    int e = blockIdx.x * blockDim.x + threadIdx.x;
    if (e >= E) return;
    __builtin_prefetch(ei + E + e + 16384, 0, 0);
    int d = (int)ei[E + e];
#pragma unroll
    for (int h = 0; h < H; ++h) {
        float ex = __expf(ebuf[e * H + h] - m[d * H + h]);
        ebuf[e * H + h] = ex;
        atomicAdd(&den[d * H + h], ex);
    }
}

// 16 threads / edge, 4 channels each (float4 gather, 4 f32 atomics to L2-resident rows)
template <int H>
__global__ void edge_agg_k(const long long* __restrict__ ei, const float* __restrict__ ebuf,
                           const float* __restrict__ den, const float* __restrict__ hfeat,
                           float* __restrict__ out, int E) {
    long long t = (long long)blockIdx.x * blockDim.x + threadIdx.x;
    if (t >= (long long)E * 16) return;
    int e = (int)(t >> 4);
    int q = (int)(t & 15);
    int c = q * 4;
    int h = (H == 1) ? 0 : (c >> 4);
    int s = (int)ei[e];
    int d = (int)ei[E + e];
    float alpha = ebuf[e * H + h] / (den[d * H + h] + EPSA);
    const float4 hv = *(const float4*)(hfeat + (size_t)s * 64 + c);
    float* o = out + (size_t)d * 64 + c;
    atomicAdd(o + 0, hv.x * alpha);
    atomicAdd(o + 1, hv.y * alpha);
    atomicAdd(o + 2, hv.z * alpha);
    atomicAdd(o + 3, hv.w * alpha);
}

// ---------------- final: relu(out2 + b2) @ Wc + bc ----------------

__global__ void finalize_k(const float* __restrict__ out2, const float* __restrict__ b2,
                           const float* __restrict__ Wc, const float* __restrict__ bc,
                           float* __restrict__ y, int n) {
    int t = blockIdx.x * blockDim.x + threadIdx.x;
    if (t >= n) return;
    const float* row = out2 + (size_t)t * 64;
    float acc = 0.f;
#pragma unroll
    for (int c = 0; c < 64; ++c) {
        float v = row[c] + b2[c];
        v = v > 0.f ? v : 0.f;
        acc += v * Wc[c];
    }
    y[t] = acc + bc[0];
}

// ---------------- launcher ----------------

extern "C" void kernel_launch(void* const* d_in, const int* in_sizes, int n_in,
                              void* d_out, int out_size, void* d_ws, size_t ws_size,
                              hipStream_t stream) {
    const float*     x      = (const float*)d_in[0];
    const long long* ei     = (const long long*)d_in[1];   // int64 [2,E]
    const float*     W1     = (const float*)d_in[2];
    const float*     a_src1 = (const float*)d_in[3];
    const float*     a_dst1 = (const float*)d_in[4];
    const float*     b1     = (const float*)d_in[5];
    const float*     W2     = (const float*)d_in[6];
    const float*     a_src2 = (const float*)d_in[7];
    const float*     a_dst2 = (const float*)d_in[8];
    const float*     b2     = (const float*)d_in[9];
    const float*     Wc     = (const float*)d_in[10];
    const float*     bc     = (const float*)d_in[11];
    float*           y      = (float*)d_out;

    char* ws = (char*)d_ws;
    // layout (bytes, all 256-aligned)
    const size_t off_xb   = 0;                              // N*128 bf16 (later reused: N*64 bf16)
    const size_t off_h1   = off_xb   + (size_t)NN * F_INN * 2;   // N*64 f32 (h1, later h2)
    const size_t off_out1 = off_h1   + (size_t)NN * HID * 4;     // N*64 f32 (out1, later out2)
    const size_t off_eb   = off_out1 + (size_t)NN * HID * 4;     // E*4 f32 (edge scratch)
    const size_t off_as   = off_eb   + (size_t)NE * 4 * 4;
    const size_t off_ad   = off_as   + (size_t)NN * 4 * 4;
    const size_t off_m    = off_ad   + (size_t)NN * 4 * 4;
    const size_t off_den  = off_m    + (size_t)NN * 4 * 4;
    const size_t off_wt   = off_den  + (size_t)NN * 4 * 4;       // 64*128 bf16 max

    unsigned short* xb   = (unsigned short*)(ws + off_xb);
    float*          h12  = (float*)(ws + off_h1);
    float*          outb = (float*)(ws + off_out1);
    float*          eb   = (float*)(ws + off_eb);
    float*          as   = (float*)(ws + off_as);
    float*          ad   = (float*)(ws + off_ad);
    float*          m    = (float*)(ws + off_m);
    float*          den  = (float*)(ws + off_den);
    unsigned short* wt   = (unsigned short*)(ws + off_wt);

    const int B = 256;
    const int gemm_blocks = (NN + 127) / 128;

    // ---------------- layer 1 (H=4, C=16, concat) ----------------
    cvt_bf16_k<<<(NN * F_INN + B - 1) / B, B, 0, stream>>>(x, xb, NN * F_INN);
    cvt_w_t_k<<<(F_INN * HID + B - 1) / B, B, 0, stream>>>(W1, wt, F_INN, HID);
    gemm_bf16_wmma<F_INN><<<gemm_blocks, B, 0, stream>>>(xb, wt, h12, NN);

    node_alpha_k<4, 16><<<(NN * 4 + B - 1) / B, B, 0, stream>>>(h12, a_src1, a_dst1, as, ad, m, den, NN);
    zero_f32_k<<<(NN * HID + B - 1) / B, B, 0, stream>>>(outb, NN * HID);

    edge_max_k<4><<<(NE + B - 1) / B, B, 0, stream>>>(ei, as, ad, eb, m, NE);
    edge_exp_k<4><<<(NE + B - 1) / B, B, 0, stream>>>(ei, eb, m, den, NE);
    edge_agg_k<4><<<(int)(((long long)NE * 16 + B - 1) / B), B, 0, stream>>>(ei, eb, den, h12, outb, NE);

    // relu(out1 + b1) -> bf16 (reuse xb region as layer-2 input)
    bias_relu_bf16_k<<<(NN * HID + B - 1) / B, B, 0, stream>>>(outb, b1, xb, NN * HID);

    // ---------------- layer 2 (H=1, C=64, mean over 1 head == identity) ----------------
    cvt_w_t_k<<<(HID * HID + B - 1) / B, B, 0, stream>>>(W2, wt, HID, HID);
    gemm_bf16_wmma<HID><<<gemm_blocks, B, 0, stream>>>(xb, wt, h12, NN);   // h2 overwrites h1

    node_alpha_k<1, 64><<<(NN + B - 1) / B, B, 0, stream>>>(h12, a_src2, a_dst2, as, ad, m, den, NN);
    zero_f32_k<<<(NN * HID + B - 1) / B, B, 0, stream>>>(outb, NN * HID); // out2 overwrites out1

    edge_max_k<1><<<(NE + B - 1) / B, B, 0, stream>>>(ei, as, ad, eb, m, NE);
    edge_exp_k<1><<<(NE + B - 1) / B, B, 0, stream>>>(ei, eb, m, den, NE);
    edge_agg_k<1><<<(int)(((long long)NE * 16 + B - 1) / B), B, 0, stream>>>(ei, eb, den, h12, outb, NE);

    finalize_k<<<(NN + B - 1) / B, B, 0, stream>>>(outb, b2, Wc, bc, y, NN);
}
